// distance_func_18030272709283
// MI455X (gfx1250) — matrix operand order repaired and checked by reading, not compiled, not run on hardware
//
#include <hip/hip_runtime.h>

typedef _Float16 v16h __attribute__((ext_vector_type(16)));
typedef _Float16 v8h  __attribute__((ext_vector_type(8)));
typedef _Float16 v4h  __attribute__((ext_vector_type(4)));
typedef float    v8f  __attribute__((ext_vector_type(8)));

#define SHUF16(a, b) __builtin_shufflevector(a, b, 0,1,2,3,4,5,6,7,8,9,10,11,12,13,14,15)

namespace {

constexpr int   kThreads   = 256;   // 8 waves of 32
constexpr int   kN         = 512;
constexpr int   kC         = 128;
constexpr int   kRowsPerWG = 128;   // 8 waves x 16 rows
constexpr float kEps       = 1e-5f;
constexpr float kSlope     = 0.01f;

// ---- shared layout for the preconverted parameter block (same in ws and LDS) ----
constexpr int OFF_W1 = 0;                       // 128x128 f16 (BN-scale folded)
constexpr int OFF_W2 = OFF_W1 + 128 * 128 * 2;  // 128x128 f16
constexpr int OFF_W3 = OFF_W2 + 128 * 128 * 2;  // 64x128 f16
constexpr int OFF_W4 = OFF_W3 + 64 * 128 * 2;   // 64x64 f16
constexpr int OFF_W5 = OFF_W4 + 64 * 64 * 2;    // 64 f16
constexpr int OFF_T1 = OFF_W5 + 64 * 2;         // 128 f32 folded biases
constexpr int OFF_T2 = OFF_T1 + 128 * 4;
constexpr int OFF_T3 = OFF_T2 + 128 * 4;        // 64 f32
constexpr int OFF_T4 = OFF_T3 + 64 * 4;         // 64 f32
constexpr int kWBytes = OFF_T4 + 64 * 4;        // 91776 B parameter block
// ---- LDS-only region ----
constexpr int OFF_XI = kWBytes;                 // 128 f32 (x row i)
constexpr int OFF_A  = OFF_XI + 128 * 4;        // 128x128 f16 activations
constexpr int OFF_B  = OFF_A + 128 * 128 * 2;   // 128x128 f16 activations
constexpr int kLdsBytes = OFF_B + 128 * 128 * 2; // 157824 B < 160 KB

struct Params {
    const float *x;
    const float *W1, *b1, *g1, *be1, *m1, *v1;
    const float *W2, *b2, *g2, *be2, *m2, *v2;
    const float *W3, *b3, *g3, *be3, *m3, *v3;
    const float *W4, *b4, *g4, *be4, *m4, *v4;
    const float *W5, *b5;
    float *out;
    unsigned char *wpre;   // preconverted parameter block (in d_ws)
};

// |a-b| rounded to f16. The abs is opaque inline asm: InstCombine's fabs idiom
// + SLP's v2f32 fabs hits a gfx1250 ISel bug in this toolchain.
__device__ inline _Float16 habs_diff(float a, float b) {
    float d = a - b;
    float r;
    asm("v_and_b32 %0, 0x7fffffff, %1" : "=v"(r) : "v"(d));
    return (_Float16)r;
}

// Convert W (cout x cin, f32) -> f16 with BN scale folded per output row.
template <int COUT, int CIN>
__device__ inline void conv_weights(const float* __restrict__ Wg,
                                    const float* __restrict__ g,
                                    const float* __restrict__ v,
                                    _Float16* __restrict__ Wd, int tid, int nthr) {
    constexpr int total4 = COUT * CIN / 4;
    const float4* src = (const float4*)Wg;
    v4h* dst = (v4h*)Wd;
    for (int q = tid; q < total4; q += nthr) {
        const int o = (4 * q) / CIN;
        const float s = g[o] * rsqrtf(v[o] + kEps);
        float4 w = src[q];
        v4h h;
        h[0] = (_Float16)(w.x * s); h[1] = (_Float16)(w.y * s);
        h[2] = (_Float16)(w.z * s); h[3] = (_Float16)(w.w * s);
        dst[q] = h;
    }
}

// Folded bias: t = (b - m)*s + beta
template <int COUT>
__device__ inline void conv_bias(const float* __restrict__ b, const float* __restrict__ g,
                                 const float* __restrict__ be, const float* __restrict__ m,
                                 const float* __restrict__ v, float* __restrict__ tl,
                                 int tid, int nthr) {
    for (int o = tid; o < COUT; o += nthr) {
        const float s = g[o] * rsqrtf(v[o] + kEps);
        tl[o] = (b[o] - m[o]) * s + be[o];
    }
}

// One fused conv+BN+LeakyReLU layer for this wave's 16 rows.
// inb:  [128 rows][CIN] f16 LDS, row-major.  Wl: [COUT][CIN] f16 LDS.
template <int CIN, int COUT>
__device__ inline void layer(const _Float16* __restrict__ inb,
                             const _Float16* __restrict__ Wl,
                             const float* __restrict__ tl,
                             _Float16* __restrict__ outb,
                             int row0, int lane16, int hi) {
    constexpr int KT = CIN / 32;
    constexpr int NT = COUT / 16;

    // A fragments (16x32 f16): lanes 0-15 -> M=lane, K = k0+0..7 & k0+16..23
    //                          lanes 16-31 -> M=lane-16, K = k0+8..15 & k0+24..31
    v16h A[KT];
    {
        const _Float16* pa = inb + (row0 + lane16) * CIN + hi * 8;
        #pragma unroll
        for (int kt = 0; kt < KT; ++kt) {
            v8h a0 = *(const v8h*)(pa + kt * 32);
            v8h a1 = *(const v8h*)(pa + kt * 32 + 16);
            A[kt] = SHUF16(a0, a1);
        }
    }

    #pragma unroll
    for (int nt = 0; nt < NT; ++nt) {
        const int col = nt * 16 + lane16;
        v8f acc = {0.f, 0.f, 0.f, 0.f, 0.f, 0.f, 0.f, 0.f};
        // B fragments (32x16): lanes 0-15 -> N=lane, K=0..7 & 16..23; lanes 16-31 -> K=8..15 & 24..31
        const _Float16* pb = Wl + col * CIN + hi * 8;
        #pragma unroll
        for (int kt = 0; kt < KT; ++kt) {
            v8h b0 = *(const v8h*)(pb + kt * 32);
            v8h b1 = *(const v8h*)(pb + kt * 32 + 16);
            v16h Bf = SHUF16(b0, b1);
            acc = __builtin_amdgcn_wmma_f32_16x16x32_f16(
                false, A[kt], false, Bf, (short)0, acc, false, false);
        }
        const float tv = tl[col];
        // C/D layout: VGPR r, lanes 0-15 -> M=r; lanes 16-31 -> M=r+8; N = lane%16
        #pragma unroll
        for (int r = 0; r < 8; ++r) {
            float val = acc[r] + tv;
            val = fmaxf(val, kSlope * val);           // LeakyReLU(0.01)
            outb[(row0 + r + hi * 8) * COUT + col] = (_Float16)val;
        }
    }
}

} // namespace

// ---- one-time parameter conversion: BN-folded f16 weights + f32 biases -> d_ws ----
__global__ __launch_bounds__(kThreads)
void prep_params(Params p) {
    unsigned char* ws = p.wpre;
    const int tid  = blockIdx.x * kThreads + threadIdx.x;
    const int nthr = gridDim.x * kThreads;
    conv_weights<128, 128>(p.W1, p.g1, p.v1, (_Float16*)(ws + OFF_W1), tid, nthr);
    conv_weights<128, 128>(p.W2, p.g2, p.v2, (_Float16*)(ws + OFF_W2), tid, nthr);
    conv_weights<64, 128>(p.W3, p.g3, p.v3, (_Float16*)(ws + OFF_W3), tid, nthr);
    conv_weights<64, 64>(p.W4, p.g4, p.v4, (_Float16*)(ws + OFF_W4), tid, nthr);
    if (tid < 64) ((_Float16*)(ws + OFF_W5))[tid] = (_Float16)p.W5[tid];
    conv_bias<128>(p.b1, p.g1, p.be1, p.m1, p.v1, (float*)(ws + OFF_T1), tid, nthr);
    conv_bias<128>(p.b2, p.g2, p.be2, p.m2, p.v2, (float*)(ws + OFF_T2), tid, nthr);
    conv_bias<64>(p.b3, p.g3, p.be3, p.m3, p.v3, (float*)(ws + OFF_T3), tid, nthr);
    conv_bias<64>(p.b4, p.g4, p.be4, p.m4, p.v4, (float*)(ws + OFF_T4), tid, nthr);
}

__global__ __launch_bounds__(kThreads, 2)
void fused_pairwise_mlp(Params p) {
    extern __shared__ __align__(16) char smem[];
    _Float16* W1h = (_Float16*)(smem + OFF_W1);
    _Float16* W2h = (_Float16*)(smem + OFF_W2);
    _Float16* W3h = (_Float16*)(smem + OFF_W3);
    _Float16* W4h = (_Float16*)(smem + OFF_W4);
    _Float16* W5h = (_Float16*)(smem + OFF_W5);
    float*    t1  = (float*)(smem + OFF_T1);
    float*    t2  = (float*)(smem + OFF_T2);
    float*    t3  = (float*)(smem + OFF_T3);
    float*    t4  = (float*)(smem + OFF_T4);
    float*    xi  = (float*)(smem + OFF_XI);
    _Float16* bufA = (_Float16*)(smem + OFF_A);
    _Float16* bufB = (_Float16*)(smem + OFF_B);

    const int tid = threadIdx.x;
    const int p0  = blockIdx.x * kRowsPerWG;      // flat (b,i,j) base; 128 rows share i
    const int bi  = p0 >> 18;                     // / (512*512)
    const int rem = p0 & ((1 << 18) - 1);
    const int irow = rem >> 9;                    // i
    const int j0   = rem & (kN - 1);              // j base (multiple of 128)

    // ---- async copy of the preconverted 91776-B parameter block into LDS ----
    {
        const unsigned long long wbase = (unsigned long long)p.wpre;
        for (int off = tid * 16; off < kWBytes; off += kThreads * 16) {
            unsigned long long ga = wbase + (unsigned)off;
            asm volatile("global_load_async_to_lds_b128 %0, %1, off"
                         :: "v"(off), "v"(ga) : "memory");
        }
    }
    if (tid < kC) xi[tid] = p.x[(size_t)(bi * kN + irow) * kC + tid];
    asm volatile("s_wait_asynccnt 0x0" ::: "memory");
    __syncthreads();

    // ---- h0 = |x_i - x_j| -> f16, staged per-wave-owned rows (no barrier needed) ----
    {
        const int row = tid >> 1;                 // wave w's threads cover rows 16w..16w+15
        const int bch = (tid & 1) * 64;
        const float* xj = p.x + (size_t)(bi * kN + j0 + row) * kC + bch;
        const float* xr = xi + bch;
        _Float16* dst = bufA + row * kC + bch;
        #pragma unroll
        for (int k = 0; k < 64; k += 8) {
            float4 a0 = *(const float4*)(xj + k);
            float4 a1 = *(const float4*)(xj + k + 4);
            float4 c0 = *(const float4*)(xr + k);
            float4 c1 = *(const float4*)(xr + k + 4);
            v8h h;
            h[0] = habs_diff(c0.x, a0.x); h[1] = habs_diff(c0.y, a0.y);
            h[2] = habs_diff(c0.z, a0.z); h[3] = habs_diff(c0.w, a0.w);
            h[4] = habs_diff(c1.x, a1.x); h[5] = habs_diff(c1.y, a1.y);
            h[6] = habs_diff(c1.z, a1.z); h[7] = habs_diff(c1.w, a1.w);
            *(v8h*)(dst + k) = h;
        }
    }

    // ---- per-wave fused layer chain (each wave owns 16 rows end-to-end) ----
    const int lane   = tid & 31;
    const int lane16 = lane & 15;
    const int hi     = lane >> 4;
    const int row0   = (tid >> 5) * 16;

    layer<128, 128>(bufA, W1h, t1, bufB, row0, lane16, hi);   // 32 WMMA
    layer<128, 128>(bufB, W2h, t2, bufA, row0, lane16, hi);   // 32 WMMA
    layer<128, 64>(bufA, W3h, t3, bufB, row0, lane16, hi);    // 16 WMMA
    layer<64, 64>(bufB, W4h, t4, bufA, row0, lane16, hi);     //  8 WMMA

    // ---- final 64 -> 1 dot product ----
    __syncthreads();
    if (tid < kRowsPerWG) {
        const v8h* hv = (const v8h*)(bufA + tid * 64);
        const v8h* wv = (const v8h*)W5h;
        float sum = 0.f;
        #pragma unroll
        for (int k = 0; k < 8; ++k) {
            v8h hh = hv[k];
            v8h ww = wv[k];
            #pragma unroll
            for (int e = 0; e < 8; ++e) sum += (float)hh[e] * (float)ww[e];
        }
        p.out[p0 + tid] = sum + p.b5[0];
    }
}

extern "C" void kernel_launch(void* const* d_in, const int* in_sizes, int n_in,
                              void* d_out, int out_size, void* d_ws, size_t ws_size,
                              hipStream_t stream) {
    (void)in_sizes; (void)n_in; (void)out_size; (void)ws_size;
    const float* const* in = (const float* const*)d_in;
    Params p;
    p.x  = in[0];
    p.W1 = in[1];  p.b1 = in[2];  p.g1 = in[3];  p.be1 = in[4];  p.m1 = in[5];  p.v1 = in[6];
    p.W2 = in[7];  p.b2 = in[8];  p.g2 = in[9];  p.be2 = in[10]; p.m2 = in[11]; p.v2 = in[12];
    p.W3 = in[13]; p.b3 = in[14]; p.g3 = in[15]; p.be3 = in[16]; p.m3 = in[17]; p.v3 = in[18];
    p.W4 = in[19]; p.b4 = in[20]; p.g4 = in[21]; p.be4 = in[22]; p.m4 = in[23]; p.v4 = in[24];
    p.W5 = in[25]; p.b5 = in[26];
    p.out  = (float*)d_out;
    p.wpre = (unsigned char*)d_ws;

    // 1) one-time (per launch) BN-folding + f16 conversion of all parameters
    hipLaunchKernelGGL(prep_params, dim3(16), dim3(kThreads), 0, stream, p);
    // 2) fused pairwise-MLP over all (b,i,j)
    const int nblocks = (2 * kN * kN) / kRowsPerWG;  // 4096
    hipLaunchKernelGGL(fused_pairwise_mlp, dim3(nblocks), dim3(kThreads),
                       kLdsBytes, stream, p);
}
